// ConvBlock_15032385536186
// MI455X (gfx1250) — matrix operand order repaired and checked by reading, not compiled, not run on hardware
//
#include <hip/hip_runtime.h>

// CDNA5 / gfx1250 wave32 WMMA types
typedef __attribute__((ext_vector_type(16))) __bf16 v16bf;
typedef __attribute__((ext_vector_type(8)))  float  v8f;

union Frag {
  uint4 u[2];   // 32 bytes = 16 bf16 elements
  v16bf v;
};

__device__ __forceinline__ unsigned short f2bf(float f) {
  unsigned int u = __float_as_uint(f);
  u += 0x7FFFu + ((u >> 16) & 1u);        // round-to-nearest-even
  return (unsigned short)(u >> 16);
}

#define Wn  256
#define Cn  256
#define CKn 32

// ---- prep kernel: transpose + convert weights fp32 -> bf16 into workspace ----
// WfT/WgT: [32][256], WhT: [256][256], all row-major [n][k]
__global__ void wconvert_kernel(const float* __restrict__ Wf,
                                const float* __restrict__ Wg,
                                const float* __restrict__ Wh,
                                unsigned short* __restrict__ WfT,
                                unsigned short* __restrict__ WgT,
                                unsigned short* __restrict__ WhT) {
  int t = blockIdx.x * blockDim.x + threadIdx.x;   // 0..65535
  int n = t >> 8;
  int k = t & 255;
  if (n < CKn) {
    WfT[n * Cn + k] = f2bf(Wf[k * CKn + n]);
    WgT[n * Cn + k] = f2bf(Wg[k * CKn + n]);
  }
  WhT[n * Cn + k] = f2bf(Wh[k * Cn + n]);
}

// ---- main kernel: one workgroup per (b,h) row-pair ----
__global__ __launch_bounds__(256)
void attn_row_kernel(const float* __restrict__ x,
                     const float* __restrict__ bfv,
                     const float* __restrict__ bgv,
                     const float* __restrict__ bhv,
                     const unsigned short* __restrict__ WfT,
                     const unsigned short* __restrict__ WgT,
                     const unsigned short* __restrict__ WhT,
                     float* __restrict__ out) {
  extern __shared__ unsigned short S[];
  unsigned short* Xbf  = S;            // [256][256] bf16  (128 KB)
  unsigned short* attn = S + 65536;    // [256][256] bf16  (128 KB)
  unsigned short* fS   = S + 131072;   // [256][32]  bf16  ( 16 KB)
  unsigned short* gS   = S + 139264;   // [256][32]  bf16  ( 16 KB)
  unsigned short* hT   = S + 147456;   // [32][256]  bf16  ( 16 KB) transposed h-chunk

  const int tid  = threadIdx.x;
  const int lane = tid & 31;
  const int wv   = tid >> 5;           // 8 waves
  const int l15  = lane & 15;
  const int hs   = lane >> 4;          // lane half select

  const long long rowblk = blockIdx.x;                 // 0..1023 = b*H+h
  const float* xp = x   + rowblk * (long long)(Wn * Cn);
  float*       op = out + rowblk * (long long)(Wn * Cn);

  // ---- Stage 1: load X (contiguous 256KB f32), convert to bf16 in LDS ----
  {
    const float4* x4 = (const float4*)xp;
    for (int i = tid; i < (Wn * Cn) / 4; i += 256) {
      float4 v = x4[i];
      uint2 p;
      p.x = (unsigned)f2bf(v.x) | ((unsigned)f2bf(v.y) << 16);
      p.y = (unsigned)f2bf(v.z) | ((unsigned)f2bf(v.w) << 16);
      *(uint2*)(Xbf + i * 4) = p;
    }
  }
  __syncthreads();

  // ---- Stage 2: f = X*Wf + bf, g = X*Wg + bg  (bf16 results in LDS) ----
  for (int t = wv; t < 64; t += 8) {
    const bool isG = (t >= 32);
    const int tt = t & 31;
    const int m0 = (tt >> 1) * 16;
    const int n0 = (tt & 1) * 16;
    const int n  = n0 + l15;
    const unsigned short* WT = isG ? WgT : WfT;
    v8f c = {};
    for (int k0 = 0; k0 < Cn; k0 += 32) {
      Frag a, b;
      const unsigned short* ar = Xbf + (m0 + l15) * Cn + k0 + hs * 8;
      a.u[0] = *(const uint4*)(ar);
      a.u[1] = *(const uint4*)(ar + 16);
      const unsigned short* br = WT + n * Cn + k0 + hs * 8;
      b.u[0] = *(const uint4*)(br);
      b.u[1] = *(const uint4*)(br + 16);
      c = __builtin_amdgcn_wmma_f32_16x16x32_bf16(false, a.v, false, b.v,
                                                  (short)0, c, false, false);
    }
    const float bias = isG ? bgv[n] : bfv[n];
    unsigned short* dst = isG ? gS : fS;
#pragma unroll
    for (int r = 0; r < 8; ++r) {
      const int M = m0 + hs * 8 + r;
      dst[M * CKn + n] = f2bf(c[r] + bias);
    }
  }
  __syncthreads();

  // ---- Stage 3: attn = sigmoid(f * g^T)  (single K=32 WMMA per tile) ----
  for (int t = wv; t < 256; t += 8) {
    const int m0 = (t >> 4) * 16;
    const int n0 = (t & 15) * 16;
    Frag a, b;
    const unsigned short* ar = fS + (m0 + l15) * CKn + hs * 8;
    a.u[0] = *(const uint4*)(ar);
    a.u[1] = *(const uint4*)(ar + 16);
    const unsigned short* br = gS + (n0 + l15) * CKn + hs * 8;
    b.u[0] = *(const uint4*)(br);
    b.u[1] = *(const uint4*)(br + 16);
    v8f c = {};
    c = __builtin_amdgcn_wmma_f32_16x16x32_bf16(false, a.v, false, b.v,
                                                (short)0, c, false, false);
#pragma unroll
    for (int r = 0; r < 8; ++r) {
      const int M = m0 + hs * 8 + r;
      const float sgm = 1.0f / (1.0f + __expf(-c[r]));
      attn[M * Cn + n0 + l15] = f2bf(sgm);
    }
  }
  __syncthreads();

  // ---- Stage 4: fused h / out, chunked over 32 output channels ----
  for (int cc = 0; cc < 8; ++cc) {
    const int nc = cc * 32;
    // h-chunk = X*Wh[:,nc:nc+32] + bh, stored TRANSPOSED ([n][k]) in LDS
    for (int t = wv; t < 32; t += 8) {
      const int m0 = (t >> 1) * 16;
      const int n0 = (t & 1) * 16;
      const int n  = nc + n0 + l15;
      v8f c = {};
      for (int k0 = 0; k0 < Cn; k0 += 32) {
        Frag a, b;
        const unsigned short* ar = Xbf + (m0 + l15) * Cn + k0 + hs * 8;
        a.u[0] = *(const uint4*)(ar);
        a.u[1] = *(const uint4*)(ar + 16);
        const unsigned short* br = WhT + n * Cn + k0 + hs * 8;
        b.u[0] = *(const uint4*)(br);
        b.u[1] = *(const uint4*)(br + 16);
        c = __builtin_amdgcn_wmma_f32_16x16x32_bf16(false, a.v, false, b.v,
                                                    (short)0, c, false, false);
      }
      const float bias = bhv[n];
      union { unsigned short s[8]; uint4 q; } pk;
#pragma unroll
      for (int r = 0; r < 8; ++r) pk.s[r] = f2bf(c[r] + bias);
      // D VGPR r covers consecutive M -> one contiguous 16B LDS store
      *(uint4*)(hT + (n0 + l15) * Cn + m0 + hs * 8) = pk.q;
    }
    __syncthreads();

    // out[:, nc:nc+32] = attn * h-chunk (f32 accumulate, straight to HBM)
    for (int t = wv; t < 32; t += 8) {
      const int m0 = (t >> 1) * 16;
      const int n0 = (t & 1) * 16;
      v8f c = {};
      for (int k0 = 0; k0 < Cn; k0 += 32) {
        Frag a, b;
        const unsigned short* ar = attn + (m0 + l15) * Cn + k0 + hs * 8;
        a.u[0] = *(const uint4*)(ar);
        a.u[1] = *(const uint4*)(ar + 16);
        const unsigned short* br = hT + (n0 + l15) * Cn + k0 + hs * 8;
        b.u[0] = *(const uint4*)(br);
        b.u[1] = *(const uint4*)(br + 16);
        c = __builtin_amdgcn_wmma_f32_16x16x32_bf16(false, a.v, false, b.v,
                                                    (short)0, c, false, false);
      }
#pragma unroll
      for (int r = 0; r < 8; ++r) {
        op[(m0 + hs * 8 + r) * Cn + nc + n0 + l15] = c[r];
      }
    }
    __syncthreads();
  }
}

extern "C" void kernel_launch(void* const* d_in, const int* in_sizes, int n_in,
                              void* d_out, int out_size, void* d_ws, size_t ws_size,
                              hipStream_t stream) {
  const float* x  = (const float*)d_in[0];
  const float* Wf = (const float*)d_in[1];
  const float* bf = (const float*)d_in[2];
  const float* Wg = (const float*)d_in[3];
  const float* bg = (const float*)d_in[4];
  const float* Wh = (const float*)d_in[5];
  const float* bh = (const float*)d_in[6];
  float* out = (float*)d_out;

  // workspace: bf16 transposed weights (16KB + 16KB + 128KB = 160KB)
  unsigned short* WfT = (unsigned short*)d_ws;
  unsigned short* WgT = WfT + 32 * 256;
  unsigned short* WhT = WgT + 32 * 256;

  wconvert_kernel<<<256, 256, 0, stream>>>(Wf, Wg, Wh, WfT, WgT, WhT);

  const size_t lds_bytes = 311296;  // 304 KB dynamic LDS (< 320 KB/WGP)
  attn_row_kernel<<<1024, 256, lds_bytes, stream>>>(x, bf, bg, bh,
                                                    WfT, WgT, WhT, out);
}
